// MeshGraphNet_54898271977900
// MI455X (gfx1250) — matrix operand order repaired and checked by reading, not compiled, not run on hardware
//
#include <hip/hip_runtime.h>

typedef _Float16 h16;
typedef __attribute__((ext_vector_type(16))) _Float16 v16h;
typedef __attribute__((ext_vector_type(8)))  _Float16 v8h;
typedef __attribute__((ext_vector_type(4)))  _Float16 v4h;
typedef __attribute__((ext_vector_type(8)))  float    v8f;
typedef __attribute__((ext_vector_type(4)))  float    v4f;
typedef __attribute__((ext_vector_type(4)))  unsigned int u32x4;
typedef __attribute__((ext_vector_type(8)))  unsigned int u32x8;

#define NN   50000
#define NE   300000
#define NL   4
#define DN_  11
#define DE_  3
#define H_   128
#define DO_  2
#define WAVES 4

// ---------------- TDM: DMA a contiguous block (n_dwords) from global to LDS -----------------
// D# group0: count=1 | lds_addr | global_addr[56:0] | type=2.  group1: data_size=4B, 1-row tile.
__device__ __forceinline__ void tdm_copy_to_lds(const void* gsrc, void* lds_dst, int n_dwords) {
    unsigned long long ga = (unsigned long long)(uintptr_t)gsrc;
    unsigned int ldsoff = (unsigned int)(uintptr_t)lds_dst;  // low 32 bits of generic = LDS offset
    u32x4 g0;
    g0[0] = 1u;                                                     // count=1, user mode
    g0[1] = ldsoff;                                                 // lds_addr (bytes)
    g0[2] = (unsigned int)ga;                                       // global_addr[31:0]
    g0[3] = ((unsigned int)(ga >> 32) & 0x01FFFFFFu) | 0x80000000u; // addr[56:32] | type=2
    unsigned int nd = (unsigned int)n_dwords;
    u32x8 g1;
    g1[0] = 2u << 16;                         // workgroup_mask=0, data_size=4B
    g1[1] = (nd & 0xFFFFu) << 16;             // tensor_dim0[15:0]
    g1[2] = ((nd >> 16) & 0xFFFFu) | (1u << 16); // tensor_dim0[31:16] | tensor_dim1=1
    g1[3] = (nd & 0xFFFFu) << 16;             // tensor_dim1 hi=0 | tile_dim0 = nd
    g1[4] = 1u;                               // tile_dim1=1, tile_dim2=0
    g1[5] = nd;                               // tensor_dim0_stride lo
    g1[6] = 0u;                               // stride hi / dim1_stride lo
    g1[7] = 0u;
    asm volatile("tensor_load_to_lds %0, %1" :: "s"(g0), "s"(g1) : "memory");
}

// ---------------- WMMA fragment helpers (CDNA5 16x16x32 f16) ----------------
// A (16xK row-major panel, stride K halves): lane&15 = row M,
// lane<16: K {0..15} of chunk; lane>=16: K {16..31}.
__device__ __forceinline__ v16h load_a_frag(const h16* rowchunk, int lane) {
    const int ko = (lane & 16) ? 16 : 0;
    return *(const v16h*)(rowchunk + ko);
}

// B (Kx16 tile) from transposed weights Wt[n][k] (row-major in n, stride K):
// lane&15 = column N, 16 contiguous K values per lane; lanes>=16 get K+16.
__device__ __forceinline__ v16h load_b_frag(const h16* wt, int K, int n, int kc, int lane) {
    const h16* p = wt + (size_t)n * K + kc * 32 + ((lane & 16) ? 16 : 0);
    return *(const v16h*)p;
}

// D = A[16xK] * Wt^T[Kx128] + bias ; acc[nt] covers output cols nt*16..nt*16+15
template <int KCH>
__device__ __forceinline__ void wave_gemm(const h16* apanel, int K, const h16* wt,
                                          const float* bias, int lane, v8f acc[8]) {
    const int n0 = lane & 15;
#pragma unroll
    for (int nt = 0; nt < 8; ++nt) {
        float b = bias[nt * 16 + n0];
#pragma unroll
        for (int r = 0; r < 8; ++r) acc[nt][r] = b;
    }
    const h16* arow = apanel + (size_t)(lane & 15) * K;
#pragma unroll
    for (int kc = 0; kc < KCH; ++kc) {
        v16h a = load_a_frag(arow + kc * 32, lane);
#pragma unroll
        for (int nt = 0; nt < 8; ++nt) {
            v16h bf = load_b_frag(wt, K, nt * 16 + n0, kc, lane);
            acc[nt] = __builtin_amdgcn_wmma_f32_16x16x32_f16(
                false, a, false, bf, (short)0, acc[nt], false, false);
        }
    }
}

// C/D layout: VGPR r, lanes0-15 -> (M=r, N=lane), lanes16-31 -> (M=r+8, N=lane-16)
__device__ __forceinline__ void store_acc_f32(float* obuf, const v8f acc[8], int lane) {
    const int mofs = (lane & 16) ? 8 : 0;
    const int n0 = lane & 15;
#pragma unroll
    for (int nt = 0; nt < 8; ++nt)
#pragma unroll
        for (int r = 0; r < 8; ++r)
            obuf[(r + mofs) * H_ + nt * 16 + n0] = acc[nt][r];
}

__device__ __forceinline__ void store_acc_relu_h16(h16* hbuf, const v8f acc[8], int lane) {
    const int mofs = (lane & 16) ? 8 : 0;
    const int n0 = lane & 15;
#pragma unroll
    for (int nt = 0; nt < 8; ++nt)
#pragma unroll
        for (int r = 0; r < 8; ++r)
            hbuf[(r + mofs) * H_ + nt * 16 + n0] = (h16)fmaxf(acc[nt][r], 0.0f);
}

// LayerNorm over H=128 of LDS buf[16][128]; optional residual RMW into dst;
// optional fused scatter-add into agg[col[row]] (sum aggregation at target nodes).
__device__ __forceinline__ void wave_ln_store(const float* obuf, const float* g, const float* be,
                                              float* dst, int lane, bool valid, bool residual,
                                              float* agg, int aggcol) {
    const int r  = lane & 15;
    const int c0 = (lane >> 4) * 64;
    const float* rowp = obuf + r * H_;
    float s = 0.0f, s2 = 0.0f;
    for (int c = c0; c < c0 + 64; ++c) { float v = rowp[c]; s += v; s2 += v * v; }
    s  += __shfl_xor(s, 16, 32);
    s2 += __shfl_xor(s2, 16, 32);
    const float mean = s * (1.0f / H_);
    const float var  = s2 * (1.0f / H_) - mean * mean;
    const float inv  = rsqrtf(var + 1e-5f);
    if (valid) {
        float* drow = dst + (size_t)r * H_;
        float* arow = agg ? (agg + (size_t)aggcol * H_) : nullptr;
        for (int c = c0; c < c0 + 64; ++c) {
            float v = (rowp[c] - mean) * inv * g[c] + be[c];
            if (residual) v += drow[c];
            drow[c] = v;
            if (agg) unsafeAtomicAdd(&arow[c], v);
        }
    }
}

// ---------------- weight prep: f32[K][128] -> f16[128][K] ----------------
__global__ void transpose_cvt_kernel(const float* __restrict__ src, h16* __restrict__ dst, int K) {
    int i = blockIdx.x * blockDim.x + threadIdx.x;
    if (i >= K * H_) return;
    int k = i / H_, n = i % H_;
    dst[(size_t)n * K + k] = (h16)src[i];
}

// ---------------- encoder: norm -> Lin(din,128) ReLU (VALU) -> Lin(128,128) (WMMA) -> LN --------
__global__ void encoder_kernel(const float* __restrict__ xin, int din,
                               const float* __restrict__ meanv, const float* __restrict__ stdv,
                               const float* __restrict__ w1, const float* __restrict__ b1,
                               const h16* __restrict__ w2t, const float* __restrict__ b2,
                               const float* __restrict__ g, const float* __restrict__ be,
                               float* __restrict__ out, int rows) {
    __shared__ __align__(32) char smem[WAVES * (16 * H_ * 2 + 16 * H_ * 4)];
    const int wave = threadIdx.x >> 5, lane = threadIdx.x & 31;
    const int tile = blockIdx.x * WAVES + wave;
    const bool valid = (tile * 16) < rows;
    const int t = valid ? tile : 0;

    h16*   h1   = (h16*)(smem + wave * 12288);
    float* obuf = (float*)(smem + wave * 12288 + 4096);

    const int r = lane & 15, c0 = (lane >> 4) * 64;
    const float* xrow = xin + (size_t)(t * 16 + r) * din;
    float xn[DN_];
    for (int d = 0; d < din; ++d) xn[d] = (xrow[d] - meanv[d]) / stdv[d];
    for (int c = c0; c < c0 + 64; ++c) {
        float s = b1[c];
        for (int d = 0; d < din; ++d) s += xn[d] * w1[d * H_ + c];
        h1[r * H_ + c] = (h16)fmaxf(s, 0.0f);
    }
    __syncthreads();

    v8f acc[8];
    wave_gemm<4>(h1, H_, w2t, b2, lane, acc);
    store_acc_f32(obuf, acc, lane);
    __syncthreads();
    wave_ln_store(obuf, g, be, out + (size_t)t * 16 * H_, lane, valid, false, nullptr, 0);
}

// ---- edge MLP: [x[row] x[col] e] (3H)->H ReLU ->H, LN, residual into e, fused agg scatter ----
__global__ void edge_mlp_kernel(const int* __restrict__ ei, const float* __restrict__ x,
                                float* __restrict__ e, float* __restrict__ agg,
                                const h16* __restrict__ w1t, const float* __restrict__ b1,
                                const h16* __restrict__ w2t, const float* __restrict__ b2,
                                const float* __restrict__ g, const float* __restrict__ be) {
    __shared__ __align__(32) char smem[WAVES * (16 * 384 * 2 + 16 * H_ * 2) + H_ * H_ * 2]; // 96KB
    const int wave = threadIdx.x >> 5, lane = threadIdx.x & 31;
    const int tile = blockIdx.x * WAVES + wave;
    const bool valid = (tile * 16) < NE;
    const int t = valid ? tile : 0;
    const int ebase = t * 16;

    h16*   apan = (h16*)(smem + wave * 16384);
    h16*   h1   = (h16*)(smem + wave * 16384 + 12288);
    h16*   wlds = (h16*)(smem + WAVES * 16384);   // block-shared GEMM2 weights [128][128]
    float* obuf = (float*)(smem + wave * 16384);  // reuse A-panel space after GEMM1

    // TDM: one DMA per block pulls w2t (32KB) into LDS; wave 0 issues and later waits.
    if (wave == 0) tdm_copy_to_lds(w2t, wlds, H_ * H_ / 2);

    // prefetch next tile's e rows (speculative; stays inside workspace)
    if (lane < 16) __builtin_prefetch(e + (size_t)(ebase + 16 + lane) * H_, 0, 0);

    // stage A panel [16][384] f16: segments x[row] | x[col] | e
    for (int q = lane; q < 16 * 96; q += 32) {
        int row = q / 96, kq = (q % 96) * 4;
        int j = ebase + row;
        const float* src;
        if (kq < 128)      src = x + (size_t)ei[j] * H_ + kq;
        else if (kq < 256) src = x + (size_t)ei[NE + j] * H_ + (kq - 128);
        else               src = e + (size_t)j * H_ + (kq - 256);
        v4f v = *(const v4f*)src;
        v4h o; o[0] = (h16)v[0]; o[1] = (h16)v[1]; o[2] = (h16)v[2]; o[3] = (h16)v[3];
        *(v4h*)(apan + row * 384 + kq) = o;
    }
    __syncthreads();

    v8f acc[8];
    wave_gemm<12>(apan, 384, w1t, b1, lane, acc);
    store_acc_relu_h16(h1, acc, lane);
    if (wave == 0) __builtin_amdgcn_s_wait_tensorcnt(0);
    __syncthreads();                               // h1 visible + TDM weights landed
    wave_gemm<4>(h1, H_, wlds, b2, lane, acc);
    store_acc_f32(obuf, acc, lane);
    __syncthreads();

    const int aggcol = ei[NE + ebase + (lane & 15)];
    wave_ln_store(obuf, g, be, e + (size_t)ebase * H_, lane, valid, true, agg, aggcol);
}

__global__ void zero_kernel(float* p, int n) {
    int i = blockIdx.x * blockDim.x + threadIdx.x;
    if (i < n) p[i] = 0.0f;
}

// ---------------- node MLP: [x agg] (2H) -> H ReLU -> H, LN, residual into x ----------------
__global__ void node_mlp_kernel(const float* __restrict__ agg, float* __restrict__ x,
                                const h16* __restrict__ w1t, const float* __restrict__ b1,
                                const h16* __restrict__ w2t, const float* __restrict__ b2,
                                const float* __restrict__ g, const float* __restrict__ be) {
    __shared__ __align__(32) char smem[WAVES * (16 * 256 * 2 + 16 * H_ * 2) + H_ * H_ * 2]; // 80KB
    const int wave = threadIdx.x >> 5, lane = threadIdx.x & 31;
    const int tile = blockIdx.x * WAVES + wave;
    const bool valid = (tile * 16) < NN;
    const int t = valid ? tile : 0;
    const int nbase = t * 16;

    h16*   apan = (h16*)(smem + wave * 12288);
    h16*   h1   = (h16*)(smem + wave * 12288 + 8192);
    h16*   wlds = (h16*)(smem + WAVES * 12288);
    float* obuf = (float*)(smem + wave * 12288);  // reuse A-panel space

    if (wave == 0) tdm_copy_to_lds(w2t, wlds, H_ * H_ / 2);

    // stage A panel [16][256] f16: segments x | agg
    for (int q = lane; q < 16 * 64; q += 32) {
        int row = q / 64, kq = (q % 64) * 4;
        const float* src = (kq < 128) ? (x + (size_t)(nbase + row) * H_ + kq)
                                      : (agg + (size_t)(nbase + row) * H_ + (kq - 128));
        v4f v = *(const v4f*)src;
        v4h o; o[0] = (h16)v[0]; o[1] = (h16)v[1]; o[2] = (h16)v[2]; o[3] = (h16)v[3];
        *(v4h*)(apan + row * 256 + kq) = o;
    }
    __syncthreads();

    v8f acc[8];
    wave_gemm<8>(apan, 256, w1t, b1, lane, acc);
    store_acc_relu_h16(h1, acc, lane);
    if (wave == 0) __builtin_amdgcn_s_wait_tensorcnt(0);
    __syncthreads();
    wave_gemm<4>(h1, H_, wlds, b2, lane, acc);
    store_acc_f32(obuf, acc, lane);
    __syncthreads();
    wave_ln_store(obuf, g, be, x + (size_t)nbase * H_, lane, valid, true, nullptr, 0);
}

// ---------------- decoder: ReLU(x@W1+b1)@W2+b2 -> [N,2] ----------------
__global__ void decoder_kernel(const float* __restrict__ x,
                               const h16* __restrict__ w1t, const float* __restrict__ b1,
                               const float* __restrict__ w2, const float* __restrict__ b2,
                               float* __restrict__ out) {
    __shared__ __align__(32) char smem[WAVES * (16 * H_ * 2 + 16 * H_ * 2)];  // 32 KB
    const int wave = threadIdx.x >> 5, lane = threadIdx.x & 31;
    const int tile = blockIdx.x * WAVES + wave;
    const bool valid = (tile * 16) < NN;
    const int t = valid ? tile : 0;

    h16* apan = (h16*)(smem + wave * 8192);
    h16* h1   = (h16*)(smem + wave * 8192 + 4096);

    for (int q = lane; q < 16 * 32; q += 32) {
        int row = q / 32, kq = (q % 32) * 4;
        v4f v = *(const v4f*)(x + (size_t)(t * 16 + row) * H_ + kq);
        v4h o; o[0] = (h16)v[0]; o[1] = (h16)v[1]; o[2] = (h16)v[2]; o[3] = (h16)v[3];
        *(v4h*)(apan + row * H_ + kq) = o;
    }
    __syncthreads();

    v8f acc[8];
    wave_gemm<4>(apan, H_, w1t, b1, lane, acc);
    store_acc_relu_h16(h1, acc, lane);
    __syncthreads();

    const int r = lane & 15, o = lane >> 4;
    float s = b2[o];
    for (int k = 0; k < H_; ++k) s += (float)h1[r * H_ + k] * w2[k * DO_ + o];
    if (valid) out[(size_t)(t * 16 + r) * DO_ + o] = s;
}

// ---------------- host launch ----------------
extern "C" void kernel_launch(void* const* d_in, const int* in_sizes, int n_in,
                              void* d_out, int out_size, void* d_ws, size_t ws_size,
                              hipStream_t stream) {
    (void)in_sizes; (void)n_in; (void)out_size; (void)ws_size;
    const float* x        = (const float*)d_in[0];
    const int*   ei       = (const int*)d_in[1];
    const float* eattr    = (const float*)d_in[2];
    const float* mean_x   = (const float*)d_in[3];
    const float* std_x    = (const float*)d_in[4];
    const float* mean_e   = (const float*)d_in[5];
    const float* std_e    = (const float*)d_in[6];
    const float* ne_w1 = (const float*)d_in[7],  *ne_b1 = (const float*)d_in[8];
    const float* ne_w2 = (const float*)d_in[9],  *ne_b2 = (const float*)d_in[10];
    const float* ne_g  = (const float*)d_in[11], *ne_be = (const float*)d_in[12];
    const float* ee_w1 = (const float*)d_in[13], *ee_b1 = (const float*)d_in[14];
    const float* ee_w2 = (const float*)d_in[15], *ee_b2 = (const float*)d_in[16];
    const float* ee_g  = (const float*)d_in[17], *ee_be = (const float*)d_in[18];
    const float* pe_w1 = (const float*)d_in[19], *pe_b1 = (const float*)d_in[20];
    const float* pe_w2 = (const float*)d_in[21], *pe_b2 = (const float*)d_in[22];
    const float* pe_g  = (const float*)d_in[23], *pe_be = (const float*)d_in[24];
    const float* pn_w1 = (const float*)d_in[25], *pn_b1 = (const float*)d_in[26];
    const float* pn_w2 = (const float*)d_in[27], *pn_b2 = (const float*)d_in[28];
    const float* pn_g  = (const float*)d_in[29], *pn_be = (const float*)d_in[30];
    const float* de_w1 = (const float*)d_in[31], *de_b1 = (const float*)d_in[32];
    const float* de_w2 = (const float*)d_in[33], *de_b2 = (const float*)d_in[34];

    const size_t nh = (size_t)NN * H_, eh = (size_t)NE * H_;
    float* x_enc = (float*)d_ws;
    float* e_enc = x_enc + nh;
    float* agg   = e_enc + eh;
    h16* ne_w2t = (h16*)(agg + nh);
    h16* ee_w2t = ne_w2t + 128 * 128;
    h16* de_w1t = ee_w2t + 128 * 128;
    h16* pe_w1t = de_w1t + 128 * 128;            // NL x [128][384]
    h16* pe_w2t = pe_w1t + (size_t)NL * 128 * 384;
    h16* pn_w1t = pe_w2t + (size_t)NL * 128 * 128;
    h16* pn_w2t = pn_w1t + (size_t)NL * 128 * 256;

    auto T = [&](const float* s, h16* d, int K) {
        int n = K * H_;
        transpose_cvt_kernel<<<(n + 255) / 256, 256, 0, stream>>>(s, d, K);
    };
    T(ne_w2, ne_w2t, 128);
    T(ee_w2, ee_w2t, 128);
    T(de_w1, de_w1t, 128);
    for (int i = 0; i < NL; ++i) {
        T(pe_w1 + (size_t)i * 384 * 128, pe_w1t + (size_t)i * 128 * 384, 384);
        T(pe_w2 + (size_t)i * 128 * 128, pe_w2t + (size_t)i * 128 * 128, 128);
        T(pn_w1 + (size_t)i * 256 * 128, pn_w1t + (size_t)i * 128 * 256, 256);
        T(pn_w2 + (size_t)i * 128 * 128, pn_w2t + (size_t)i * 128 * 128, 128);
    }

    const int NT_N = NN / 16, NB_N = (NT_N + WAVES - 1) / WAVES;  // 3125 tiles, 782 blocks
    const int NT_E = NE / 16, NB_E = (NT_E + WAVES - 1) / WAVES;  // 18750 tiles, 4688 blocks

    encoder_kernel<<<NB_N, WAVES * 32, 0, stream>>>(x, DN_, mean_x, std_x, ne_w1, ne_b1,
                                                    ne_w2t, ne_b2, ne_g, ne_be, x_enc, NN);
    encoder_kernel<<<NB_E, WAVES * 32, 0, stream>>>(eattr, DE_, mean_e, std_e, ee_w1, ee_b1,
                                                    ee_w2t, ee_b2, ee_g, ee_be, e_enc, NE);
    for (int i = 0; i < NL; ++i) {
        zero_kernel<<<((int)nh + 255) / 256, 256, 0, stream>>>(agg, (int)nh);
        edge_mlp_kernel<<<NB_E, WAVES * 32, 0, stream>>>(
            ei, x_enc, e_enc, agg,
            pe_w1t + (size_t)i * 128 * 384, pe_b1 + (size_t)i * H_,
            pe_w2t + (size_t)i * 128 * 128, pe_b2 + (size_t)i * H_,
            pe_g + (size_t)i * H_, pe_be + (size_t)i * H_);
        node_mlp_kernel<<<NB_N, WAVES * 32, 0, stream>>>(
            agg, x_enc,
            pn_w1t + (size_t)i * 128 * 256, pn_b1 + (size_t)i * H_,
            pn_w2t + (size_t)i * 128 * 128, pn_b2 + (size_t)i * H_,
            pn_g + (size_t)i * H_, pn_be + (size_t)i * H_);
    }
    decoder_kernel<<<NB_N, WAVES * 32, 0, stream>>>(x_enc, de_w1t, de_b1, de_w2, de_b2,
                                                    (float*)d_out);
}
